// AttentionWrapper_12111807774776
// MI455X (gfx1250) — compile-verified, matrix operand order checked by reading, not run on hardware
//
#include <hip/hip_runtime.h>

// ---- problem constants ----
#define BB 2
#define HH 64
#define WW 64
#define CC 256
#define HD 32
#define NHEADS 8
#define KW 7
#define CW 22              // candidate strip width: 16 + (7-1)
#define NC 154             // 7 * 22 candidates
#define NCP 160            // padded to multiple of 32
#define QKVN 768

typedef __attribute__((ext_vector_type(16))) __bf16 v16bf;
typedef __attribute__((ext_vector_type(8)))  __bf16 bf16x8;
typedef __attribute__((ext_vector_type(8)))  float  v8f;

union ABu { v16bf v; bf16x8 h[2]; __bf16 e[16]; };

__device__ __forceinline__ unsigned short f2bf(float f) {
  union { float f; unsigned u; } v; v.f = f;
  unsigned r = (v.u + 0x7FFFu + ((v.u >> 16) & 1u)) >> 16;
  return (unsigned short)r;
}

// ---- K0a: x (B,C,H,W) f32 -> xb (B,H,W,C) bf16 ----
__global__ __launch_bounds__(256) void cast_x_kernel(const float* __restrict__ x,
                                                     unsigned short* __restrict__ xb) {
  int idx = blockIdx.x * 256 + threadIdx.x;          // exactly 2*256*4096 threads
  int pix = idx & 4095;
  int c   = (idx >> 12) & 255;
  int b   = idx >> 20;
  xb[(size_t)((b << 12) + pix) * CC + c] = f2bf(x[idx]);
}

// ---- K0b: cast qkv_w (768x256) and proj_w (256x256) to bf16 ----
__global__ __launch_bounds__(256) void cast_w_kernel(const float* __restrict__ qw,
                                                     const float* __restrict__ pw,
                                                     unsigned short* __restrict__ wq,
                                                     unsigned short* __restrict__ wp) {
  int idx = blockIdx.x * 256 + threadIdx.x;          // exactly 262144 threads
  if (idx < 196608) wq[idx] = f2bf(qw[idx]);
  else              wp[idx - 196608] = f2bf(pw[idx - 196608]);
}

// ---- K1: qkv = xb @ qkv_w^T + b ; q pre-scaled by hd^-0.5 ; bf16 out ----
__global__ __launch_bounds__(256) void qkv_gemm_kernel(
    const unsigned short* __restrict__ xb, const unsigned short* __restrict__ wq,
    const float* __restrict__ qkv_b, unsigned short* __restrict__ qkv) {
  int lane = threadIdx.x & 31;
  int wid  = threadIdx.x >> 5;
  int t  = blockIdx.x * 8 + wid;                     // 24576 tiles
  int mt = t / 48, nt = t - mt * 48;
  int m0 = mt * 16, n0 = nt * 16;
  int rw = lane & 15;                                // A row / B column / D column
  int o  = (lane >> 4) * 8;                          // K sub-slot per lane half
  int ncol = n0 + rw;
  const unsigned short* ap = xb + (size_t)(m0 + rw) * CC + o;
  const unsigned short* bp = wq + (size_t)ncol * CC + o;
  v8f acc = {};
  for (int kc = 0; kc < 8; ++kc) {
    ABu a, b;
    a.h[0] = *(const bf16x8*)(ap + kc * 32);
    a.h[1] = *(const bf16x8*)(ap + kc * 32 + 16);
    b.h[0] = *(const bf16x8*)(bp + kc * 32);
    b.h[1] = *(const bf16x8*)(bp + kc * 32 + 16);
    acc = __builtin_amdgcn_wmma_f32_16x16x32_bf16(false, a.v, false, b.v,
                                                  (short)0, acc, false, false);
  }
  float bias  = qkv_b[ncol];
  float scale = (ncol < 256) ? 0.17677669529663687f : 1.0f;  // hd^-0.5 for q
  int half = lane >> 4;
  for (int j = 0; j < 8; ++j) {
    int m = m0 + j + 8 * half;
    qkv[(size_t)m * QKVN + ncol] = f2bf((acc[j] + bias) * scale);
  }
}

// ---- K2: neighborhood attention, one wave = (b, i, 16 pixels in j, head) ----
#define WPB 2
#define PERWAVE_LDS 25600  // 10240 (V stage) + 10240 (logits f32) + 5120 (attn bf16)

__global__ __launch_bounds__(64) void attn_kernel(
    const unsigned short* __restrict__ qkv, const float* __restrict__ rpb,
    unsigned short* __restrict__ ao) {
  extern __shared__ char smem[];
  int lane = threadIdx.x & 31;
  int wid  = threadIdx.x >> 5;
  char* base = smem + wid * PERWAVE_LDS;
  unsigned short* vst = (unsigned short*)base;               // [160][32] bf16
  float*          att = (float*)(base + 10240);              // [16][160] f32
  unsigned short* a16 = (unsigned short*)(base + 20480);     // [16][160] bf16

  int tile = blockIdx.x * WPB + wid;                         // 4096 tiles
  int h  = tile & 7;
  int jt = (tile >> 3) & 3;
  int i  = (tile >> 5) & 63;
  int b  = tile >> 11;
  int j0 = jt * 16;
  int si = min(max(i - 3, 0), HH - KW);
  int cb = min(max(j0 - 3, 0), WW - KW);
  int mrow = (b * HH + i) * WW;

  // stage V candidates into LDS (160 vectors x 32 bf16)
  for (int idx = lane; idx < NCP * 4; idx += 32) {
    int c = idx >> 2, part = idx & 3;
    int cc = c < NC ? c : NC - 1;
    int r = cc / CW, tcol = cc - r * CW;
    int col = min(cb + tcol, WW - 1);
    int mc = (b * HH + si + r) * WW + col;
    *(uint4*)(vst + c * HD + part * 8) =
        *(const uint4*)(qkv + (size_t)mc * QKVN + 512 + h * HD + part * 8);
  }

  int o  = (lane >> 4) * 8;
  int rw = lane & 15;

  // Q in A-operand layout (16 pixels x 32 dims)
  ABu aq;
  {
    const unsigned short* qp = qkv + (size_t)(mrow + j0 + rw) * QKVN + h * HD + o;
    aq.h[0] = *(const bf16x8*)qp;
    aq.h[1] = *(const bf16x8*)(qp + 16);
  }

  // logits = Q x K^T over 160 candidates: 10 WMMAs -> LDS
  for (int ntile = 0; ntile < 10; ++ntile) {
    int c = ntile * 16 + rw;
    int cc = c < NC ? c : NC - 1;
    int r = cc / CW, tcol = cc - r * CW;
    int col = min(cb + tcol, WW - 1);
    int mc = (b * HH + si + r) * WW + col;
    ABu bk;
    const unsigned short* kp = qkv + (size_t)mc * QKVN + 256 + h * HD + o;
    bk.h[0] = *(const bf16x8*)kp;
    bk.h[1] = *(const bf16x8*)(kp + 16);
    v8f acc = {};
    acc = __builtin_amdgcn_wmma_f32_16x16x32_bf16(false, aq.v, false, bk.v,
                                                  (short)0, acc, false, false);
    int half = lane >> 4;
    for (int j = 0; j < 8; ++j)
      att[(j + 8 * half) * NCP + ntile * 16 + rw] = acc[j];
  }

  // masked bias + softmax; lanes 0-15 each own one pixel row (re-converges below)
  if (lane < 16) {
    int p  = lane;
    int j  = j0 + p;
    int sj = min(max(j - 3, 0), WW - KW);
    const float* bias = rpb + h * 169;
    float mx = -3.0e38f;
    for (int r = 0; r < KW; ++r) {
      int reli = si + r - i + 6;
      for (int tc = 0; tc < CW; ++tc) {
        int colj = cb + tc;
        int cidx = r * CW + tc;
        float l = -3.0e38f;
        if (colj >= sj && colj <= sj + 6)
          l = att[p * NCP + cidx] + bias[reli * 13 + (colj - j + 6)];
        att[p * NCP + cidx] = l;
        mx = fmaxf(mx, l);
      }
    }
    float sum = 0.f;
    for (int cidx = 0; cidx < NC; ++cidx)
      sum += __expf(att[p * NCP + cidx] - mx);
    float inv = 1.0f / sum;
    for (int cidx = 0; cidx < NC; ++cidx)
      a16[p * NCP + cidx] = f2bf(__expf(att[p * NCP + cidx] - mx) * inv);
    for (int cidx = NC; cidx < NCP; ++cidx)
      a16[p * NCP + cidx] = 0;
  }

  // out = attn(16x160) x V(160x32): 2 N-tiles x 5 K-chunks = 10 WMMAs
  for (int dt = 0; dt < 2; ++dt) {
    int dcol = dt * 16 + rw;
    v8f acc = {};
    for (int kc = 0; kc < 5; ++kc) {
      ABu aa, bb;
      aa.h[0] = *(const bf16x8*)(a16 + rw * NCP + kc * 32 + o);
      aa.h[1] = *(const bf16x8*)(a16 + rw * NCP + kc * 32 + o + 16);
      for (int e2 = 0; e2 < 8; ++e2) {
        bb.e[e2]     = *(const __bf16*)(vst + (kc * 32 + o + e2) * HD + dcol);
        bb.e[8 + e2] = *(const __bf16*)(vst + (kc * 32 + o + 16 + e2) * HD + dcol);
      }
      acc = __builtin_amdgcn_wmma_f32_16x16x32_bf16(false, aa.v, false, bb.v,
                                                    (short)0, acc, false, false);
    }
    int half = lane >> 4;
    for (int j = 0; j < 8; ++j) {
      int m = mrow + j0 + j + 8 * half;
      ao[(size_t)m * CC + h * HD + dcol] = f2bf(acc[j]);
    }
  }
}

// ---- K3: out = x + (ao @ proj_w^T + b), transposed back to (B,C,H,W) ----
__global__ __launch_bounds__(256) void proj_gemm_kernel(
    const unsigned short* __restrict__ ao, const unsigned short* __restrict__ wp,
    const float* __restrict__ proj_b, const float* __restrict__ x,
    float* __restrict__ out) {
  int lane = threadIdx.x & 31;
  int wid  = threadIdx.x >> 5;
  int t  = blockIdx.x * 8 + wid;                     // 8192 tiles
  int mt = t >> 4, nt = t & 15;
  int m0 = mt * 16, n0 = nt * 16;
  int rw = lane & 15;
  int o  = (lane >> 4) * 8;
  int ncol = n0 + rw;
  const unsigned short* ap = ao + (size_t)(m0 + rw) * CC + o;
  const unsigned short* bp = wp + (size_t)ncol * CC + o;
  v8f acc = {};
  for (int kc = 0; kc < 8; ++kc) {
    ABu a, b;
    a.h[0] = *(const bf16x8*)(ap + kc * 32);
    a.h[1] = *(const bf16x8*)(ap + kc * 32 + 16);
    b.h[0] = *(const bf16x8*)(bp + kc * 32);
    b.h[1] = *(const bf16x8*)(bp + kc * 32 + 16);
    acc = __builtin_amdgcn_wmma_f32_16x16x32_bf16(false, a.v, false, b.v,
                                                  (short)0, acc, false, false);
  }
  float bias = proj_b[ncol];
  int half = lane >> 4;
  for (int j = 0; j < 8; ++j) {
    int m   = m0 + j + 8 * half;
    int b_  = m >> 12;
    int pix = m & 4095;
    size_t addr = (size_t)b_ * (CC * 4096) + (size_t)ncol * 4096 + pix;
    out[addr] = x[addr] + acc[j] + bias;
  }
}

extern "C" void kernel_launch(void* const* d_in, const int* in_sizes, int n_in,
                              void* d_out, int out_size, void* d_ws, size_t ws_size,
                              hipStream_t stream) {
  (void)in_sizes; (void)n_in; (void)out_size; (void)ws_size;
  const float* x      = (const float*)d_in[0];
  const float* qkv_w  = (const float*)d_in[1];
  const float* qkv_b  = (const float*)d_in[2];
  const float* proj_w = (const float*)d_in[3];
  const float* proj_b = (const float*)d_in[4];
  const float* rpb    = (const float*)d_in[5];
  float* out = (float*)d_out;

  // workspace partitions (total 21,495,808 bytes, all 256B-aligned)
  char* ws = (char*)d_ws;
  unsigned short* xb   = (unsigned short*)(ws);                 // 4,194,304
  unsigned short* wq   = (unsigned short*)(ws + 4194304);       //   393,216
  unsigned short* wp   = (unsigned short*)(ws + 4587520);       //   131,072
  unsigned short* qkv  = (unsigned short*)(ws + 4718592);       // 12,582,912
  unsigned short* ao   = (unsigned short*)(ws + 17301504);      // 4,194,304

  cast_x_kernel  <<<8192, 256, 0, stream>>>(x, xb);
  cast_w_kernel  <<<1024, 256, 0, stream>>>(qkv_w, proj_w, wq, wp);
  qkv_gemm_kernel<<<3072, 256, 0, stream>>>(xb, wq, qkv_b, qkv);
  attn_kernel    <<<2048, 64, WPB * PERWAVE_LDS, stream>>>(qkv, rpb, ao);
  proj_gemm_kernel<<<1024, 256, 0, stream>>>(ao, wp, proj_b, x, out);
}
